// GCN_22170621182205
// MI455X (gfx1250) — matrix operand order repaired and checked by reading, not compiled, not run on hardware
//
#include <hip/hip_runtime.h>
#include <hip/hip_bf16.h>

typedef __attribute__((ext_vector_type(8)))  _Float16 v8h;   // 16 B
typedef __attribute__((ext_vector_type(16))) _Float16 v16h;  // 32 B (A/B fragment)
typedef __attribute__((ext_vector_type(8)))  float    v8f;   // C/D fragment

// ---------------------------------------------------------------------------
// Degree / normalization kernels
// ---------------------------------------------------------------------------
__global__ void gcn_deg_init(float* __restrict__ deg, int n) {
    int i = blockIdx.x * blockDim.x + threadIdx.x;
    if (i < n) deg[i] = 1.0f;  // self-loop contribution
}

__global__ void gcn_deg_count(const int* __restrict__ col, float* __restrict__ deg, int e) {
    int i = blockIdx.x * blockDim.x + threadIdx.x;
    if (i < e) atomicAdd(&deg[col[i]], 1.0f);
}

__global__ void gcn_deg_rsqrt(float* __restrict__ deg, int n) {
    int i = blockIdx.x * blockDim.x + threadIdx.x;
    if (i < n) deg[i] = rsqrtf(deg[i]);  // deg >= 1 always (self loop)
}

// ---------------------------------------------------------------------------
// Cast activations fp32 -> f16 row-major (optionally fused ReLU).
// ---------------------------------------------------------------------------
__global__ void gcn_f16cast(const float* __restrict__ A, _Float16* __restrict__ O,
                            int total, int relu) {
    int i = blockIdx.x * blockDim.x + threadIdx.x;
    if (i < total) {
        float v = A[i];
        if (relu) v = fmaxf(v, 0.0f);
        O[i] = (_Float16)v;
    }
}

// ---------------------------------------------------------------------------
// Pack W[K,NC] fp32 -> Wt[NPAD,K] f16 (transposed, zero-padded columns).
// ---------------------------------------------------------------------------
__global__ void gcn_pack_w(const float* __restrict__ W, _Float16* __restrict__ Wt,
                           int K, int NC, int NPAD) {
    int idx = blockIdx.x * blockDim.x + threadIdx.x;  // over NPAD*K
    if (idx < NPAD * K) {
        int n = idx / K;
        int k = idx - n * K;
        Wt[idx] = (n < NC) ? (_Float16)W[(size_t)k * NC + n] : (_Float16)0.0f;
    }
}

// ---------------------------------------------------------------------------
// WMMA GEMM:  H[M,NC] = A[M,K] @ W[K,NC], A as row-major f16, W pre-packed
// transposed f16. One wave computes 16 rows x NTILES*16 cols; all fragment
// loads are contiguous b128s per the ISA 7.12.2 16-bit layouts:
//   A frag (lane r,h):  K in [8h,8h+8) and [16+8h,16+8h+8)   -> 2x16B
//   B frag (lane n,h):  K in [8h,8h+8) and [16+8h,16+8h+8)   -> 2x16B
// ---------------------------------------------------------------------------
template <int NTILES, int K, int NC>
__global__ void gcn_gemm_wmma(const _Float16* __restrict__ Af16,
                              const _Float16* __restrict__ Wt,
                              float* __restrict__ H, int M) {
    const int m0   = blockIdx.x * 16;
    const int lane = threadIdx.x & 31;  // wave32
    const int half = lane >> 4;
    const int r    = lane & 15;
    const int row  = min(m0 + r, M - 1);  // clamp: no exec-mask branches on loads

    const _Float16* __restrict__ arow = Af16 + (size_t)row * K;
    const _Float16* __restrict__ wrow = Wt + (size_t)r * K;  // + t*16*K per tile

    v8f acc[NTILES];
#pragma unroll
    for (int t = 0; t < NTILES; ++t) acc[t] = v8f{};

#pragma unroll 1
    for (int ks = 0; ks < K / 32; ++ks) {
        const _Float16* ab = arow + ks * 32 + 8 * half;
        v8h a0 = *(const v8h*)(ab);
        v8h a1 = *(const v8h*)(ab + 16);
        v16h a;
#pragma unroll
        for (int i = 0; i < 8; ++i) { a[i] = a0[i]; a[8 + i] = a1[i]; }

#pragma unroll
        for (int t = 0; t < NTILES; ++t) {
            const _Float16* bb = wrow + (size_t)t * 16 * K + ks * 32 + 8 * half;
            v8h b0 = *(const v8h*)(bb);
            v8h b1 = *(const v8h*)(bb + 16);
            v16h b;
#pragma unroll
            for (int i = 0; i < 8; ++i) { b[i] = b0[i]; b[8 + i] = b1[i]; }
            acc[t] = __builtin_amdgcn_wmma_f32_16x16x32_f16(
                false, a, false, b, (short)0, acc[t], false, false);
        }
    }

    // ---- Store D: lane -> column n, VGPR v -> row m0 + v + 8*half ----
    const int mbase = m0 + 8 * half;
    if (m0 + 16 <= M) {  // wave-uniform fast path
#pragma unroll
        for (int t = 0; t < NTILES; ++t) {
            const int n = t * 16 + r;
            if ((NC % 16 == 0) || (n < NC)) {
                float* __restrict__ hp = H + (size_t)mbase * NC + n;
#pragma unroll
                for (int v = 0; v < 8; ++v) hp[(size_t)v * NC] = acc[t][v];
            }
        }
    } else {
#pragma unroll
        for (int t = 0; t < NTILES; ++t) {
            const int n = t * 16 + r;
            if ((NC % 16 == 0) || (n < NC)) {
#pragma unroll
                for (int v = 0; v < 8; ++v)
                    if (mbase + v < M) H[(size_t)(mbase + v) * NC + n] = acc[t][v];
            }
        }
    }
}

// ---------------------------------------------------------------------------
// OUT[i,c] = dinv[i]^2 * H[i,c] + bias[c]  (self-loop + bias; initializes OUT)
// ---------------------------------------------------------------------------
__global__ void gcn_selfbias(const float* __restrict__ H,
                             const float* __restrict__ dinv,
                             const float* __restrict__ bias,
                             float* __restrict__ OUT, int total, int C) {
    int idx = blockIdx.x * blockDim.x + threadIdx.x;
    if (idx < total) {
        int n = idx / C;
        int c = idx - n * C;
        float di = dinv[n];
        OUT[idx] = di * di * H[idx] + bias[c];
    }
}

// ---------------------------------------------------------------------------
// Edge scatter-add: one wave32 per edge; lanes stride channels (coalesced
// 128B gather of H[row], f32 atomics into OUT[col]).
// ---------------------------------------------------------------------------
__global__ void gcn_scatter(const float* __restrict__ H,
                            const int* __restrict__ rowi,
                            const int* __restrict__ coli,
                            const float* __restrict__ dinv,
                            float* __restrict__ OUT, int E, int C) {
    const int wid  = (blockIdx.x * blockDim.x + threadIdx.x) >> 5;
    const int lane = threadIdx.x & 31;
    if (wid >= E) return;
    const int s = rowi[wid];
    const int d = coli[wid];
    const float nrm = dinv[s] * dinv[d];
    const float* __restrict__ hs = H + (size_t)s * C;
    float* __restrict__ od = OUT + (size_t)d * C;
    for (int c = lane; c < C; c += 32) {
        atomicAdd(&od[c], hs[c] * nrm);
    }
}

// ---------------------------------------------------------------------------
// Host-side orchestration
// ---------------------------------------------------------------------------
extern "C" void kernel_launch(void* const* d_in, const int* in_sizes, int n_in,
                              void* d_out, int out_size, void* d_ws, size_t ws_size,
                              hipStream_t stream) {
    const float* x   = (const float*)d_in[0];
    const int*  eidx = (const int*)  d_in[1];
    const float* W1  = (const float*)d_in[2];
    const float* b1  = (const float*)d_in[3];
    const float* W2  = (const float*)d_in[4];
    const float* b2  = (const float*)d_in[5];
    const float* W3  = (const float*)d_in[6];
    const float* b3  = (const float*)d_in[7];

    const int CIN = 128, CH = 128, COUT = 40, COUTP = 48;
    const int N = in_sizes[0] / CIN;
    const int E = in_sizes[1] / 2;
    const int* erow = eidx;       // edge_index[0] : sources
    const int* ecol = eidx + E;   // edge_index[1] : targets

    // Workspace: ws0[N*128]f32 | ws1[N*128]f32 | dinv[N]f32 | af16[N*128]f16 | wt[128*128]f16
    float*    ws0  = (float*)d_ws;
    float*    ws1  = ws0 + (size_t)N * CH;
    float*    dinv = ws1 + (size_t)N * CH;
    _Float16* af16 = (_Float16*)(dinv + N);
    _Float16* wt   = af16 + (size_t)N * CH;

    const int TPB = 256;
    const int mblocks = (N + 15) / 16;
    const int NK = N * CH;

    // --- degree / normalization ---
    gcn_deg_init <<<(N + TPB - 1) / TPB, TPB, 0, stream>>>(dinv, N);
    gcn_deg_count<<<(E + TPB - 1) / TPB, TPB, 0, stream>>>(ecol, dinv, E);
    gcn_deg_rsqrt<<<(N + TPB - 1) / TPB, TPB, 0, stream>>>(dinv, N);

    // --- layer 1: H = x @ W1 ; agg -> ws1 ---
    gcn_f16cast<<<(NK + TPB - 1) / TPB, TPB, 0, stream>>>(x, af16, NK, 0);
    gcn_pack_w <<<(CH * CIN + TPB - 1) / TPB, TPB, 0, stream>>>(W1, wt, CIN, CH, CH);
    gcn_gemm_wmma<8, 128, 128><<<mblocks, 32, 0, stream>>>(af16, wt, ws0, N);
    gcn_selfbias<<<(NK + TPB - 1) / TPB, TPB, 0, stream>>>(ws0, dinv, b1, ws1, NK, CH);
    gcn_scatter<<<(E + 7) / 8, TPB, 0, stream>>>(ws0, erow, ecol, dinv, ws1, E, CH);

    // --- layer 2: H = relu(agg1) @ W2 ; agg -> ws1 ---
    gcn_f16cast<<<(NK + TPB - 1) / TPB, TPB, 0, stream>>>(ws1, af16, NK, 1);
    gcn_pack_w <<<(CH * CH + TPB - 1) / TPB, TPB, 0, stream>>>(W2, wt, CH, CH, CH);
    gcn_gemm_wmma<8, 128, 128><<<mblocks, 32, 0, stream>>>(af16, wt, ws0, N);
    gcn_selfbias<<<(NK + TPB - 1) / TPB, TPB, 0, stream>>>(ws0, dinv, b2, ws1, NK, CH);
    gcn_scatter<<<(E + 7) / 8, TPB, 0, stream>>>(ws0, erow, ecol, dinv, ws1, E, CH);

    // --- layer 3: H = relu(agg2) @ W3 ; aggregate straight into d_out ---
    float* out = (float*)d_out;
    gcn_f16cast<<<(NK + TPB - 1) / TPB, TPB, 0, stream>>>(ws1, af16, NK, 1);
    gcn_pack_w <<<(COUTP * CH + TPB - 1) / TPB, TPB, 0, stream>>>(W3, wt, CH, COUT, COUTP);
    gcn_gemm_wmma<3, 128, 40><<<mblocks, 32, 0, stream>>>(af16, wt, ws0, N);
    gcn_selfbias<<<(N * COUT + TPB - 1) / TPB, TPB, 0, stream>>>(ws0, dinv, b3, out, N * COUT, COUT);
    gcn_scatter<<<(E + 7) / 8, TPB, 0, stream>>>(ws0, erow, ecol, dinv, out, E, COUT);
}
